// ParallelFFNMoE_61160334295259
// MI455X (gfx1250) — matrix operand order repaired and checked by reading, not compiled, not run on hardware
//
#include <hip/hip_runtime.h>
#include <math.h>
#include <type_traits>

typedef __bf16 bf16;
typedef __attribute__((ext_vector_type(16))) __bf16 v16bf;
typedef __attribute__((ext_vector_type(8)))  float  v8f;
typedef __attribute__((ext_vector_type(4)))  float  f32x4;
typedef __attribute__((ext_vector_type(4)))  unsigned int u32x4;
typedef __attribute__((ext_vector_type(2)))  unsigned int u32x2;

#define BM 128
#define BN 128
#define BK 32
#define LDSW 40   // bf16 per LDS row: 32 data + 8 pad -> 80B rows, 16B-aligned

__device__ __forceinline__ float gelu_tanh(float x) {
  const float c0 = 0.7978845608028654f;   // sqrt(2/pi)
  const float c1 = 0.044715f;
  return 0.5f * x * (1.0f + tanhf(c0 * (x + c1 * x * x * x)));
}

// A fragment (16x32 bf16). ISA layout: lanes 0-15 hold row m with K {0..7,16..23},
// lanes 16-31 hold row m with K {8..15,24..31}. LDS is row-major [m][k].
__device__ __forceinline__ v16bf load_fragA(const bf16* lds, int mbase, int lane) {
  const int row  = mbase + (lane & 15);
  const int koff = (lane >> 4) * 8;
  const bf16* p = lds + row * LDSW + koff;
  union { v16bf v; u32x4 q[2]; } f;
  f.q[0] = *(const u32x4*)(p);        // K koff .. koff+7
  f.q[1] = *(const u32x4*)(p + 16);   // K koff+16 .. koff+23
  return f.v;
}

// B fragment (32x16 bf16). ISA layout: lanes 0-15 hold col n with K {0..7,16..23},
// lanes 16-31 hold col n with K {8..15,24..31}. LDS holds B transposed: [n][k].
__device__ __forceinline__ v16bf load_fragB(const bf16* lds, int nbase, int lane) {
  const int col  = nbase + (lane & 15);
  const int koff = (lane >> 4) * 8;
  const bf16* p = lds + col * LDSW + koff;
  union { v16bf v; u32x4 q[2]; } f;
  f.q[0] = *(const u32x4*)(p);        // K koff .. koff+7
  f.q[1] = *(const u32x4*)(p + 16);   // K koff+16 .. koff+23
  return f.v;
}

// MODE 0: C(bf16) = gelu(A@B + bias)           (hidden activations)
// MODE 1: C(f32)  = A@B + bias                 (dense FFN output, full write)
// MODE 2: C(f32) += 0.5*wvec[e]*(A@B + bias)   only where (m & smask) >= *idp
template <typename AT, int MODE>
__global__ __launch_bounds__(256) void gemm_wmma_kernel(
    const AT* __restrict__ A, int lda,
    const float* __restrict__ Bw, int ldb,
    const float* __restrict__ bias,
    void* __restrict__ Cout, int ldc,
    int K, int smask,
    const int* __restrict__ idp,
    const float* __restrict__ wvec, int e)
{
  __shared__ bf16 ldsA[2][BM * LDSW];
  __shared__ bf16 ldsB[2][BN * LDSW];

  const int m0 = blockIdx.y * BM;
  const int n0 = blockIdx.x * BN;

  int id = 0;
  if (idp) {
    id = *idp;
    // whole 128-token block unmasked -> no MoE contribution needed, skip.
    if ((m0 & smask) + BM <= id) return;
  }

  const int tid  = threadIdx.x;
  const int lane = tid & 31;
  const int wid  = tid >> 5;           // 8 waves
  const int wm   = (wid >> 2) * 64;    // wave M offset: 0 / 64
  const int wn   = (wid & 3) * 32;     // wave N offset: 0/32/64/96

  const v8f vzero = {0.f,0.f,0.f,0.f,0.f,0.f,0.f,0.f};
  v8f acc[4][2];
  #pragma unroll
  for (int i = 0; i < 4; ++i) { acc[i][0] = vzero; acc[i][1] = vzero; }

  const int ar = tid >> 3;          // A stage: rows ar + 32*p
  const int ac = (tid & 7) * 4;     // A stage: k col, 4 elems
  // B stage: each thread owns one tile column and 16 consecutive k-values.
  // Lanes of a wave read adjacent columns at the same k -> coalesced global
  // loads; each thread's 16 k-values are contiguous in the [n][k] LDS layout
  // -> two aligned ds_store_b128 instead of a b16 scatter.
  const int bcol = tid & 127;          // 0..127
  const int bkh  = (tid >> 7) * 16;    // 0 or 16

  // Register-buffered pipeline: next tile is fetched into ra/rbv while the
  // current tile (already in LDS) feeds the WMMAs. LDS is double-buffered so
  // only one barrier per k-step is needed.
  using ARegT = typename std::conditional<sizeof(AT) == 4, f32x4, u32x2>::type;
  ARegT ra[4];
  float rbv[16];

  const AT*    aPtr = A  + (size_t)(m0 + ar) * lda + ac;
  const float* bPtr = Bw + (size_t)bkh * ldb + (n0 + bcol);

  auto load_tile = [&]() {
    #pragma unroll
    for (int p = 0; p < 4; ++p)
      ra[p] = *(const ARegT*)(aPtr + (size_t)(p * 32) * lda);
    #pragma unroll
    for (int i = 0; i < 16; ++i)
      rbv[i] = bPtr[(size_t)i * ldb];
    aPtr += BK;
    bPtr += (size_t)BK * ldb;
  };

  auto store_tile = [&](int buf) {
    #pragma unroll
    for (int p = 0; p < 4; ++p) {
      union { bf16 h[4]; u32x2 u; } t;
      if constexpr (sizeof(AT) == 4) {
        t.h[0] = (bf16)ra[p].x; t.h[1] = (bf16)ra[p].y;
        t.h[2] = (bf16)ra[p].z; t.h[3] = (bf16)ra[p].w;
      } else {
        t.u = ra[p];
      }
      *(u32x2*)(&ldsA[buf][(ar + p * 32) * LDSW + ac]) = t.u;
    }
    union { bf16 h[16]; u32x4 q[2]; } tb;
    #pragma unroll
    for (int i = 0; i < 16; ++i) tb.h[i] = (bf16)rbv[i];
    bf16* db = &ldsB[buf][bcol * LDSW + bkh];
    *(u32x4*)(db)     = tb.q[0];   // K bkh   .. bkh+7
    *(u32x4*)(db + 8) = tb.q[1];   // K bkh+8 .. bkh+15
  };

  load_tile();
  int buf = 0;
  for (int k0 = 0; k0 < K; k0 += BK) {
    store_tile(buf);
    __syncthreads();
    if (k0 + BK < K) {
      load_tile();                       // issue next global loads, no wait yet
      __builtin_prefetch(aPtr, 0, 1);    // hint tile after next into L2
      __builtin_prefetch(bPtr, 0, 1);
    }

    v16bf af[4], bg[2];
    #pragma unroll
    for (int i = 0; i < 4; ++i) af[i] = load_fragA(ldsA[buf], wm + 16 * i, lane);
    #pragma unroll
    for (int j = 0; j < 2; ++j) bg[j] = load_fragB(ldsB[buf], wn + 16 * j, lane);

    #pragma unroll
    for (int i = 0; i < 4; ++i)
      #pragma unroll
      for (int j = 0; j < 2; ++j)
        acc[i][j] = __builtin_amdgcn_wmma_f32_16x16x32_bf16(
            false, af[i], false, bg[j], (short)0, acc[i][j], false, false);

    buf ^= 1;
  }

  float scale = 1.0f;
  if (wvec) scale = 0.5f * wvec[e];

  // C/D layout: VGPR p -> M = p (lanes 0-15) or 8+p (lanes 16-31); N = lane%16
  const int cn    = lane & 15;
  const int rbase = (lane >> 4) * 8;
  #pragma unroll
  for (int j = 0; j < 2; ++j) {
    const int n  = n0 + wn + 16 * j + cn;
    const float bv = bias ? bias[n] : 0.0f;
    #pragma unroll
    for (int i = 0; i < 4; ++i) {
      const int mrow = m0 + wm + 16 * i + rbase;
      #pragma unroll
      for (int p = 0; p < 8; ++p) {
        const int m = mrow + p;
        float v = acc[i][j][p] + bv;
        if constexpr (MODE == 0) {
          ((bf16*)Cout)[(size_t)m * ldc + n] = (bf16)gelu_tanh(v);
        } else if constexpr (MODE == 1) {
          ((float*)Cout)[(size_t)m * ldc + n] = v;
        } else {
          if ((m & smask) >= id) {
            float* o = (float*)Cout + (size_t)m * ldc + n;
            *o += scale * v;
          }
        }
      }
    }
  }
}

extern "C" void kernel_launch(void* const* d_in, const int* in_sizes, int n_in,
                              void* d_out, int out_size, void* d_ws, size_t ws_size,
                              hipStream_t stream)
{
  (void)in_sizes; (void)n_in; (void)out_size; (void)ws_size;
  const float* x   = (const float*)d_in[0];
  const float* W1  = (const float*)d_in[1];
  const float* b1  = (const float*)d_in[2];
  const float* W2  = (const float*)d_in[3];
  const float* b2  = (const float*)d_in[4];
  const float* eW1 = (const float*)d_in[5];
  const float* eb1 = (const float*)d_in[6];
  const float* eW2 = (const float*)d_in[7];
  const float* eb2 = (const float*)d_in[8];
  const float* wts = (const float*)d_in[9];
  const int*   idp = (const int*)d_in[10];

  const int Bb = 2, S = 2048, D = 1024, F = 4096, E = 8;
  const int M = Bb * S;                 // 4096 tokens
  const int smask = S - 1;              // S is a power of two

  bf16* H = (bf16*)d_ws;                // M x F bf16 = 32 MB scratch, reused

  dim3 blk(256, 1, 1);
  dim3 g1(F / BN, M / BM, 1);           // N = F GEMMs
  dim3 g2(D / BN, M / BM, 1);           // N = D GEMMs

  // Dense FFN over all tokens
  gemm_wmma_kernel<float, 0><<<g1, blk, 0, stream>>>(
      x, D, W1, F, b1, H, F, D, smask, nullptr, nullptr, 0);
  gemm_wmma_kernel<bf16, 1><<<g2, blk, 0, stream>>>(
      H, F, W2, D, b2, d_out, D, F, smask, nullptr, nullptr, 0);

  // MoE experts, masked accumulate (blocks fully below `id` early-exit)
  for (int e = 0; e < E; ++e) {
    gemm_wmma_kernel<float, 0><<<g1, blk, 0, stream>>>(
        x, D, eW1 + (size_t)e * D * F, F, eb1 + (size_t)e * F,
        H, F, D, smask, idp, nullptr, 0);
    gemm_wmma_kernel<bf16, 2><<<g2, blk, 0, stream>>>(
        H, F, eW2 + (size_t)e * F * D, D, eb2 + (size_t)e * D,
        d_out, D, F, smask, idp, wts, e);
  }
}
